// SelfAttentionV2_68118181314924
// MI455X (gfx1250) — matrix operand order repaired and checked by reading, not compiled, not run on hardware
//
#include <hip/hip_runtime.h>
#include <hip/hip_fp16.h>

// ---------------------------------------------------------------------------
// Self-attention (B=8, S=4096, D=1024, single wide head) for gfx1250 / MI455X.
// f16 WMMA (v_wmma_f32_16x16x32_f16) everywhere, f32 accumulation.
// Workspace layout (f16):
//   Wt  : 3 * 1024*1024          (Wq^T, Wk^T, Wv^T)
//   Q   : 8*4096*1024  row-major, pre-scaled by 1/sqrt(1024)
//   K   : 8*4096*1024  row-major
//   Vt  : 8*1024*4096  transposed per batch  [b][e][s]
// Total ~198 MB of d_ws.
//
// Roofline: ~756 GFLOP total vs ~330 MB (k1) + ~260 MB (k2) HBM traffic
// -> compute-bound; all matmul work on v_wmma_f32_16x16x32_f16 with
// independent accumulator groups for WMMA-issue ILP.
// ---------------------------------------------------------------------------

#define DM   1024
#define SS   4096
#define BB   8

typedef __attribute__((ext_vector_type(16))) _Float16 v16h;
typedef __attribute__((ext_vector_type(8)))  float    v8f;

union FragU { uint4 u[2]; v16h h; };        // 32B: one WMMA A/B fragment per lane
union Pack8 { _Float16 h[8]; uint4 u; };    // 16B: 8 packed halves

__device__ __forceinline__ v8f wmma16(v16h a, v16h b, v8f c) {
  // D = A(16x32,f16) * B(32x16,f16) + C(16x16,f32)
  return __builtin_amdgcn_wmma_f32_16x16x32_f16(
      /*neg_a=*/false, a, /*neg_b=*/false, b,
      /*c_mod=*/(short)0, c, /*reuse_a=*/false, /*reuse_b=*/false);
}

// --------------------- kernel 0: W -> W^T (f16), LDS transpose --------------
// 64x64 tiles; coalesced reads and writes; pad to dodge bank conflicts.
__global__ __launch_bounds__(256) void prep_wt(
    const float* __restrict__ Wq, const float* __restrict__ Wk,
    const float* __restrict__ Wv, _Float16* __restrict__ Wt) {
  __shared__ _Float16 tile[64 * 65];
  const int z = blockIdx.z;
  const float* W = (z == 0) ? Wq : ((z == 1) ? Wk : Wv);
  _Float16* o = Wt + (size_t)z * DM * DM;

  const int d0 = blockIdx.x * 64;   // input row block   (d index)
  const int e0 = blockIdx.y * 64;   // input col block   (e index)
  const int tc = threadIdx.x & 63;  // column within tile
  const int tr = threadIdx.x >> 6;  // 0..3

  // read W[d0+row][e0+col], coalesced along e
#pragma unroll
  for (int i = 0; i < 16; ++i) {
    const int row = i * 4 + tr;
    tile[row * 65 + tc] = (_Float16)W[(size_t)(d0 + row) * DM + e0 + tc];
  }
  __syncthreads();
  // write Wt[e0+row][d0+col], coalesced along d
#pragma unroll
  for (int i = 0; i < 16; ++i) {
    const int row = i * 4 + tr;
    o[(size_t)(e0 + row) * DM + d0 + tc] = tile[tc * 65 + row];
  }
}

// ------------------- kernel 1: QKV projection GEMM (WMMA) -------------------
// C[32768 x 1024] = x[32768 x 1024] * W  (x3 matrices).
// Grid: x-dim = n_tile + 8*mat (24, FAST) so 24 consecutive blocks share one
// 128-row x tile through L2; y-dim = m_tile (256).
// Block tile 128x128, 8 waves in 4(M) x 2(N); wave tile 32x64 = 2x4 WMMA tiles.
__global__ __launch_bounds__(256) void qkv_gemm(
    const float* __restrict__ x, const _Float16* __restrict__ Wt,
    _Float16* __restrict__ Qm, _Float16* __restrict__ Km,
    _Float16* __restrict__ Vt) {
  __shared__ _Float16 lds[128 * 136];                 // padded, 34816 B

  const int tid  = threadIdx.x;
  const int lane = tid & 31;
  const int wave = tid >> 5;
  const int hi   = lane >> 4;          // lane half (K-split of fragments)
  const int lm   = lane & 15;          // row (A) / col (B) within tile
  const int wm   = wave >> 1;          // 0..3
  const int wn   = wave & 1;           // 0..1
  const int mat  = blockIdx.x >> 3;    // 0=Q 1=K 2=V
  const int nBlk = (blockIdx.x & 7) * 128;
  const int mBlk = blockIdx.y * 128;
  const _Float16* W = Wt + (size_t)mat * (DM * DM);

  const int mW = mBlk + wm * 32;
  const int nW = nBlk + wn * 64;

  v8f zero = {};
  v8f acc[2][4];
  for (int i = 0; i < 2; ++i)
    for (int j = 0; j < 4; ++j) acc[i][j] = zero;

  for (int k0 = 0; k0 < DM; k0 += 32) {
    // A fragments from x (f32 -> f16 on the fly); lane<16: K {0..7,16..23}
    v16h a[2];
#pragma unroll
    for (int mt = 0; mt < 2; ++mt) {
      const float* px = x + (size_t)(mW + mt * 16 + lm) * DM + k0 + hi * 8;
      float4 f0 = ((const float4*)px)[0];
      float4 f1 = ((const float4*)px)[1];
      float4 f2 = ((const float4*)(px + 16))[0];
      float4 f3 = ((const float4*)(px + 16))[1];
      v16h t;
      t[0]=(_Float16)f0.x;  t[1]=(_Float16)f0.y;  t[2]=(_Float16)f0.z;  t[3]=(_Float16)f0.w;
      t[4]=(_Float16)f1.x;  t[5]=(_Float16)f1.y;  t[6]=(_Float16)f1.z;  t[7]=(_Float16)f1.w;
      t[8]=(_Float16)f2.x;  t[9]=(_Float16)f2.y;  t[10]=(_Float16)f2.z; t[11]=(_Float16)f2.w;
      t[12]=(_Float16)f3.x; t[13]=(_Float16)f3.y; t[14]=(_Float16)f3.z; t[15]=(_Float16)f3.w;
      a[mt] = t;
    }
    // B fragments from W^T: lane holds K {8h..8h+7, 16+8h..16+8h+7} of column nW+nt*16+lm
#pragma unroll
    for (int nt = 0; nt < 4; ++nt) {
      const _Float16* pw = W + (size_t)(nW + nt * 16 + lm) * DM + k0 + hi * 8;
      FragU fb;
      fb.u[0] = ((const uint4*)pw)[0];
      fb.u[1] = ((const uint4*)(pw + 16))[0];
      acc[0][nt] = wmma16(a[0], fb.h, acc[0][nt]);
      acc[1][nt] = wmma16(a[1], fb.h, acc[1][nt]);
    }
  }

  if (mat < 2) {
    // Q (scaled by 1/sqrt(1024)) and K: row-major f16, coalesced via LDS bounce
    const float sc = (mat == 0) ? 0.03125f : 1.0f;
#pragma unroll
    for (int mt = 0; mt < 2; ++mt)
#pragma unroll
      for (int nt = 0; nt < 4; ++nt) {
        v8f c = acc[mt][nt];
        const int cc = wn * 64 + nt * 16 + lm;
        const int rbase = wm * 32 + mt * 16 + hi * 8;
#pragma unroll
        for (int r = 0; r < 8; ++r)
          lds[(rbase + r) * 136 + cc] = (_Float16)(c[r] * sc);
      }
    __syncthreads();
    _Float16* outp = (mat == 0) ? Qm : Km;
    const int row = tid >> 1, seg = tid & 1;
    const uint4* src = (const uint4*)(lds + row * 136 + seg * 64);
    uint4* dst = (uint4*)(outp + (size_t)(mBlk + row) * DM + nBlk + seg * 64);
#pragma unroll
    for (int i = 0; i < 8; ++i) dst[i] = src[i];
  } else {
    // V stored transposed [b][e][s]: each lane packs 8 consecutive s -> 16B store
#pragma unroll
    for (int mt = 0; mt < 2; ++mt)
#pragma unroll
      for (int nt = 0; nt < 4; ++nt) {
        v8f c = acc[mt][nt];
        Pack8 p;
#pragma unroll
        for (int r = 0; r < 8; ++r) p.h[r] = (_Float16)c[r];
        const int e  = nW + nt * 16 + lm;
        const int bs = mW + mt * 16 + hi * 8;
        const int bb = bs >> 12;            // S = 4096
        const int s0 = bs & (SS - 1);
        *(uint4*)(Vt + ((size_t)(bb * DM + e)) * SS + s0) = p.u;
      }
  }
}

// --------------------- kernel 2: attention (LDS-resident P) -----------------
// One workgroup = 16 queries of one batch. 8 waves.
//   LDS: P[16][4096] f16 (128KB) + Q[16][1024] f16 (32KB) + reduction scratch.
// Phase 1: wave w -> keys [512w, 512w+512): 4 groups of 8 key-tiles with
//          independent accumulators (WMMA ILP); A-fragment shared per k-step.
// Phase 2: cooperative softmax rows (max, exp, sum; 1/sum kept for epilogue).
// Phase 3: wave w -> output dims [128w, 128w+128): O = P V via WMMA.
__global__ __launch_bounds__(256) void attn(
    const _Float16* __restrict__ Qm, const _Float16* __restrict__ Km,
    const _Float16* __restrict__ Vt, float* __restrict__ out) {
  extern __shared__ char smem[];
  _Float16* P      = (_Float16*)smem;                       // 16*4096 f16
  _Float16* Qs     = (_Float16*)(smem + 131072);            // 16*1024 f16
  float*    scr    = (float*)(smem + 131072 + 32768);       // 256
  float*    rowMax = scr + 256;                             // 16
  float*    rowInv = rowMax + 16;                           // 16

  const int tid  = threadIdx.x;
  const int lane = tid & 31;
  const int wave = tid >> 5;
  const int hi   = lane >> 4;
  const int lm   = lane & 15;
  const int b    = blockIdx.y;
  const int q0   = blockIdx.x * 16;

  // ---- phase 0: copy Q tile (16x1024 f16, contiguous) into LDS ----
  {
    const uint4* src = (const uint4*)(Qm + ((size_t)(b * SS + q0)) * DM);
    uint4* dst = (uint4*)Qs;
    for (int i = tid; i < (16 * DM) / 8; i += 256) dst[i] = src[i];
  }
  __syncthreads();

  // ---- phase 1: scores -> P (f16), 8 concurrent key-tiles per group ----
  const int keyBase = wave * 512;
  for (int g = 0; g < 4; ++g) {
    const int kb = keyBase + g * 128;
    v8f zero = {};
    v8f acc[8];
#pragma unroll
    for (int t = 0; t < 8; ++t) acc[t] = zero;

    for (int k0 = 0; k0 < DM; k0 += 32) {
      FragU fa;   // A: Q rows from LDS (shared by all 8 tiles)
      fa.u[0] = *(const uint4*)(Qs + lm * DM + k0 + hi * 8);
      fa.u[1] = *(const uint4*)(Qs + lm * DM + k0 + 16 + hi * 8);
#pragma unroll
      for (int t = 0; t < 8; ++t) {
        const _Float16* krow =
            Km + ((size_t)(b * SS + kb + t * 16 + lm)) * DM + k0 + hi * 8;
        FragU fb;   // B: K^T -> key row contiguous in d
        fb.u[0] = ((const uint4*)krow)[0];
        fb.u[1] = ((const uint4*)(krow + 16))[0];
        acc[t] = wmma16(fa.h, fb.h, acc[t]);
      }
    }
#pragma unroll
    for (int t = 0; t < 8; ++t)
#pragma unroll
      for (int r = 0; r < 8; ++r)
        P[(size_t)(r + hi * 8) * SS + kb + t * 16 + lm] = (_Float16)acc[t][r];
  }
  __syncthreads();

  // ---- phase 2: softmax over each of the 16 rows (4096 keys) ----
  {
    const int row = tid & 15;       // 16 rows
    const int chunk = tid >> 4;     // 16 chunks of 256 keys
    _Float16* pr = P + (size_t)row * SS + chunk * 256;

    float m = -3.0e38f;
    for (int j = 0; j < 256; j += 8) {
      Pack8 v; v.u = *(const uint4*)(pr + j);
#pragma unroll
      for (int e = 0; e < 8; ++e) m = fmaxf(m, (float)v.h[e]);
    }
    scr[tid] = m;
    __syncthreads();
    if (tid < 16) {
      float mm = scr[tid];
      for (int c2 = 1; c2 < 16; ++c2) mm = fmaxf(mm, scr[c2 * 16 + tid]);
      rowMax[tid] = mm;
    }
    __syncthreads();

    const float M = rowMax[row];
    float sum = 0.f;
    for (int j = 0; j < 256; j += 8) {
      Pack8 v; v.u = *(const uint4*)(pr + j);
      Pack8 o;
#pragma unroll
      for (int e = 0; e < 8; ++e) {
        float p = __expf((float)v.h[e] - M);
        sum += p;
        o.h[e] = (_Float16)p;
      }
      *(uint4*)(pr + j) = o.u;    // in-place: unnormalized probs (f16)
    }
    scr[tid] = sum;
    __syncthreads();
    if (tid < 16) {
      float ss = 0.f;
      for (int c2 = 0; c2 < 16; ++c2) ss += scr[c2 * 16 + tid];
      rowInv[tid] = 1.0f / ss;
    }
    __syncthreads();
  }

  // ---- phase 3: O = P @ V  (wave owns 128 output dims, 8 accumulators) ----
  const int eBase = wave * 128;
  v8f zero = {};
  v8f acc[8];
#pragma unroll
  for (int t = 0; t < 8; ++t) acc[t] = zero;

  for (int k0 = 0; k0 < SS; k0 += 32) {
    FragU fa;   // A: probability rows from LDS (shared by all 8 tiles)
    fa.u[0] = *(const uint4*)(P + (size_t)lm * SS + k0 + hi * 8);
    fa.u[1] = *(const uint4*)(P + (size_t)lm * SS + k0 + 16 + hi * 8);
#pragma unroll
    for (int t = 0; t < 8; ++t) {
      const _Float16* vrow =
          Vt + ((size_t)(b * DM + eBase + t * 16 + lm)) * SS + k0 + hi * 8;
      FragU fb;   // B: V^T -> dim row contiguous in s
      fb.u[0] = ((const uint4*)vrow)[0];
      fb.u[1] = ((const uint4*)(vrow + 16))[0];
      acc[t] = wmma16(fa.h, fb.h, acc[t]);
    }
  }

  float inv[8];
#pragma unroll
  for (int r = 0; r < 8; ++r) inv[r] = rowInv[r + hi * 8];

#pragma unroll
  for (int t = 0; t < 8; ++t) {
    const int col = eBase + t * 16 + lm;
#pragma unroll
    for (int r = 0; r < 8; ++r) {
      const int mrow = r + hi * 8;
      out[((size_t)(b * SS + q0 + mrow)) * DM + col] = acc[t][r] * inv[r];
    }
  }
}

// ------------------------------- launcher -----------------------------------
extern "C" void kernel_launch(void* const* d_in, const int* in_sizes, int n_in,
                              void* d_out, int out_size, void* d_ws, size_t ws_size,
                              hipStream_t stream) {
  const float* x  = (const float*)d_in[0];
  const float* Wq = (const float*)d_in[1];
  const float* Wk = (const float*)d_in[2];
  const float* Wv = (const float*)d_in[3];
  float* out = (float*)d_out;

  _Float16* Wt = (_Float16*)d_ws;                       // 3 * 1024*1024
  _Float16* Qm = Wt + (size_t)3 * DM * DM;              // 8*4096*1024
  _Float16* Km = Qm + (size_t)BB * SS * DM;             // 8*4096*1024
  _Float16* Vt = Km + (size_t)BB * SS * DM;             // 8*1024*4096

  prep_wt<<<dim3(DM / 64, DM / 64, 3), 256, 0, stream>>>(Wq, Wk, Wv, Wt);

  // n+mat in the fast grid dim: 24 consecutive blocks share one x row-tile in L2
  qkv_gemm<<<dim3(24, (BB * SS) / 128), 256, 0, stream>>>(x, Wt, Qm, Km, Vt);

  const size_t smem = 131072 + 32768 + (256 + 16 + 16) * sizeof(float);
  attn<<<dim3(SS / 16, BB), 256, smem, stream>>>(Qm, Km, Vt, out);
}
